// SCLSTM_91122026152528
// MI455X (gfx1250) — matrix operand order repaired and checked by reading, not compiled, not run on hardware
//
#include <hip/hip_runtime.h>
#include <hip/hip_bf16.h>
#include <math.h>

// Problem constants
#define B_SZ 512
#define T_SZ 55
#define C_SZ 512
#define D_SZ 512
#define K_SZ 64
#define OUT_SZ 512
#define XCOLS 2048            // C + 3*D
#define XCT   (XCOLS / 32)    // 64 column tiles in X
#define HCT   (3 * D_SZ / 32) // 48 column tiles in Hprev
#define ALPHA_C (1.0f / 3.0f)

typedef __bf16 bf16_t;
typedef __bf16 v16bf __attribute__((ext_vector_type(16)));
typedef float  v8f   __attribute__((ext_vector_type(8)));

// ---------------------------------------------------------------------------
// Swizzled layouts matching CDNA5 WMMA fragment VGPR maps (ISA 7.12.2).
// A-fragment (16x32 bf16): lane = half*16 + m,  K(e) = (e>>3)*16 + half*8 + (e&7)
// Tile storage: [row_tile][col_tile][lane*16 + e], 512 bf16 per tile.
__device__ __forceinline__ size_t xswz_idx(int row, int col) {
    int rt = row >> 4, m = row & 15;
    int ct = col >> 5, kk = col & 31;
    int e    = ((kk >> 4) << 3) | (kk & 7);
    int half = (kk >> 3) & 1;
    int lane = (half << 4) | m;
    return ((size_t)(rt * XCT + ct) << 9) + (lane << 4) + e;
}
// B-fragment (32x16 bf16): lane = half*16 + n, K(e) = (e>>3)*16 + half*8 + (e&7)
// Tile storage: [k_tile][n_tile][lane*16 + e]
__device__ __forceinline__ size_t bswz_idx(int k, int n, int n_tiles) {
    int kt = k >> 5, kk = k & 31;
    int half = (kk >> 3) & 1, e = ((kk >> 4) << 3) | (kk & 7);
    int lane = (half << 4) | (n & 15);
    int nt = n >> 4;
    return ((size_t)(kt * n_tiles + nt) << 9) + (lane << 4) + e;
}

// ---------------------------------------------------------------------------
// Big-tile bf16 WMMA GEMM for the gate matmuls (N % 128 == 0, M % 128 == 0,
// (K/32) even). Block: 256 threads = 8 waves (4 row x 2 col); wave = 32x64
// (2 A-frags x 4 B-frags -> 8 WMMAs per K-step; 0.75KB L2 traffic per WMMA).
// Software-pipelined with two ping-pong operand sets.
__global__ __launch_bounds__(256)
void sclstm_gemm_big(const bf16_t* __restrict__ A, int a_ld_ct,
                     const bf16_t* __restrict__ Bw, int n_tiles,
                     float* __restrict__ C, int ldc, int Kdim)
{
    const int lane = threadIdx.x & 31;
    const int wave = threadIdx.x >> 5;
    const int wrow = wave >> 1, wcol = wave & 1;
    const int m_tile0 = blockIdx.y * 8 + wrow * 2;       // 2 consecutive m-tiles
    const int n_base  = blockIdx.x * 128 + wcol * 64;
    const int nt0 = n_base >> 4;

    const bf16_t* arow0 = A + ((size_t)m_tile0 * a_ld_ct << 9) + (lane << 4);
    const bf16_t* arow1 = arow0 + ((size_t)a_ld_ct << 9);
    const bf16_t* bbase = Bw + ((size_t)nt0 << 9) + (lane << 4);
    const size_t  bstep = (size_t)n_tiles << 9;
    const int ktc = Kdim >> 5;

    v8f acc[8];
#pragma unroll
    for (int i = 0; i < 8; ++i) acc[i] = (v8f){0.f,0.f,0.f,0.f,0.f,0.f,0.f,0.f};

    v16bf a0[2], a1[2], b0[4], b1[4];

    auto LD = [&](v16bf* a, v16bf* b, int kt) {
        a[0] = *(const v16bf*)(arow0 + ((size_t)kt << 9));
        a[1] = *(const v16bf*)(arow1 + ((size_t)kt << 9));
        const bf16_t* bp = bbase + (size_t)kt * bstep;
        b[0] = *(const v16bf*)(bp);
        b[1] = *(const v16bf*)(bp + 512);
        b[2] = *(const v16bf*)(bp + 1024);
        b[3] = *(const v16bf*)(bp + 1536);
    };
    auto MM = [&](const v16bf* a, const v16bf* b) {
#pragma unroll
        for (int j = 0; j < 4; ++j)
#pragma unroll
            for (int i = 0; i < 2; ++i)
                acc[j * 2 + i] = __builtin_amdgcn_wmma_f32_16x16x32_bf16(
                    false, a[i], false, b[j], (short)0, acc[j * 2 + i], false, false);
    };

    LD(a0, b0, 0);
    int kt = 0;
    for (; kt + 2 < ktc; kt += 2) {
        LD(a1, b1, kt + 1);
        MM(a0, b0);
        LD(a0, b0, kt + 2);
        MM(a1, b1);
    }
    LD(a1, b1, kt + 1);   // ktc even
    MM(a0, b0);
    MM(a1, b1);

    const int half = lane >> 4, nl = lane & 15;
#pragma unroll
    for (int j = 0; j < 4; ++j) {
        int col = n_base + j * 16 + nl;
#pragma unroll
        for (int i = 0; i < 2; ++i) {
            int row0 = ((m_tile0 + i) << 4) + (half << 3);
#pragma unroll
            for (int r = 0; r < 8; ++r)
                C[(size_t)(row0 + r) * ldc + col] = acc[j * 2 + i][r];
        }
    }
}

// ---------------------------------------------------------------------------
// Standard bf16 WMMA GEMM (projection): N % 128 == 0, M % 64 == 0, (K/32) even.
// Block: 256 threads = 8 waves (4 row x 2 col); wave = 16x64; block = 64x128.
__global__ __launch_bounds__(256)
void sclstm_gemm_main(const bf16_t* __restrict__ A, int a_ld_ct,
                      const bf16_t* __restrict__ Bw, int n_tiles,
                      float* __restrict__ C, int ldc,
                      const float* __restrict__ bias,
                      int Kdim)
{
    const int lane = threadIdx.x & 31;
    const int wave = threadIdx.x >> 5;
    const int wrow = wave >> 1, wcol = wave & 1;
    const int m_tile = blockIdx.y * 4 + wrow;
    const int n_base = blockIdx.x * 128 + wcol * 64;
    const int nt0 = n_base >> 4;

    const bf16_t* arow  = A + ((size_t)m_tile * a_ld_ct << 9) + (lane << 4);
    const bf16_t* bbase = Bw + ((size_t)nt0 << 9) + (lane << 4);
    const size_t  bstep = (size_t)n_tiles << 9;
    const int ktc = Kdim >> 5;

    v8f acc[4];
#pragma unroll
    for (int i = 0; i < 4; ++i) acc[i] = (v8f){0.f,0.f,0.f,0.f,0.f,0.f,0.f,0.f};

    v16bf a0, a1, b0[4], b1[4];

    auto LD = [&](v16bf& a, v16bf* b, int kt) {
        a = *(const v16bf*)(arow + ((size_t)kt << 9));
        const bf16_t* bp = bbase + (size_t)kt * bstep;
        b[0] = *(const v16bf*)(bp);
        b[1] = *(const v16bf*)(bp + 512);
        b[2] = *(const v16bf*)(bp + 1024);
        b[3] = *(const v16bf*)(bp + 1536);
    };
    auto MM = [&](const v16bf& a, const v16bf* b) {
#pragma unroll
        for (int j = 0; j < 4; ++j)
            acc[j] = __builtin_amdgcn_wmma_f32_16x16x32_bf16(
                false, a, false, b[j], (short)0, acc[j], false, false);
    };

    LD(a0, b0, 0);
    int kt = 0;
    for (; kt + 2 < ktc; kt += 2) {
        LD(a1, b1, kt + 1);
        MM(a0, b0);
        LD(a0, b0, kt + 2);
        MM(a1, b1);
    }
    LD(a1, b1, kt + 1);   // ktc even
    MM(a0, b0);
    MM(a1, b1);

    const int half = lane >> 4, nl = lane & 15;
    const int row0 = (m_tile << 4) + (half << 3);
#pragma unroll
    for (int j = 0; j < 4; ++j) {
        int col = n_base + j * 16 + nl;
        float badd = bias ? bias[col] : 0.0f;
#pragma unroll
        for (int r = 0; r < 8; ++r)
            C[(size_t)(row0 + r) * ldc + col] = acc[j][r] + badd;
    }
}

// ---------------------------------------------------------------------------
// gr GEMM: N = 64 fixed (4 n-tiles, guardless). A rows come from X for
// k-tiles [0, ktc1) and from Hprev for k-tiles [ktc1, ktc1+ktc2).
// Block: 256 threads = 8 waves; wave = one 16-row tile; grid 4 blocks.
__global__ __launch_bounds__(256)
void sclstm_gemm_gr(const bf16_t* __restrict__ X, const bf16_t* __restrict__ Hp,
                    int ktc1, int ktc2,
                    const bf16_t* __restrict__ Bw, float* __restrict__ C)
{
    const int lane = threadIdx.x & 31;
    const int wave = threadIdx.x >> 5;
    const int m_tile = blockIdx.x * 8 + wave;

    const bf16_t* arow1 = X  + ((size_t)m_tile * XCT << 9) + (lane << 4);
    const bf16_t* arow2 = Hp + ((size_t)m_tile * HCT << 9) + (lane << 4);

    v8f acc[4];
#pragma unroll
    for (int i = 0; i < 4; ++i) acc[i] = (v8f){0.f,0.f,0.f,0.f,0.f,0.f,0.f,0.f};

    for (int kt = 0; kt < ktc1; ++kt) {
        v16bf a = *(const v16bf*)(arow1 + ((size_t)kt << 9));
        const bf16_t* bp = Bw + ((size_t)(kt * 4) << 9) + (lane << 4);
#pragma unroll
        for (int j = 0; j < 4; ++j) {
            v16bf b = *(const v16bf*)(bp + ((size_t)j << 9));
            acc[j] = __builtin_amdgcn_wmma_f32_16x16x32_bf16(
                false, a, false, b, (short)0, acc[j], false, false);
        }
    }
    for (int kt = 0; kt < ktc2; ++kt) {
        v16bf a = *(const v16bf*)(arow2 + ((size_t)kt << 9));
        const bf16_t* bp = Bw + ((size_t)((ktc1 + kt) * 4) << 9) + (lane << 4);
#pragma unroll
        for (int j = 0; j < 4; ++j) {
            v16bf b = *(const v16bf*)(bp + ((size_t)j << 9));
            acc[j] = __builtin_amdgcn_wmma_f32_16x16x32_bf16(
                false, a, false, b, (short)0, acc[j], false, false);
        }
    }

    const int half = lane >> 4, nl = lane & 15;
    const int row0 = (m_tile << 4) + (half << 3);
#pragma unroll
    for (int j = 0; j < 4; ++j) {
        int col = j * 16 + nl;
#pragma unroll
        for (int r = 0; r < 8; ++r)
            C[(size_t)(row0 + r) * K_SZ + col] = acc[j][r];
    }
}

// ---------------------------------------------------------------------------
// fp32 -> swizzled bf16 conversion of x_t into X columns [0, C)
__global__ void sclstm_conv_x(const float* __restrict__ in, bf16_t* __restrict__ X, int t)
{
    int idx = blockIdx.x * blockDim.x + threadIdx.x;
    if (idx >= B_SZ * C_SZ) return;
    int b = idx / C_SZ, cc = idx % C_SZ;
    float v = in[((size_t)b * T_SZ + t) * C_SZ + cc];
    X[xswz_idx(b, cc)] = (bf16_t)v;
}

// Stacked gate weight [wx ; wh] -> swizzled bf16 (K = fin + D, N = 4D)
__global__ void sclstm_conv_gatew(const float* __restrict__ wx, const float* __restrict__ wh,
                                  bf16_t* __restrict__ dst, int fin)
{
    const int N = 4 * D_SZ;
    const int Ktot = fin + D_SZ;
    int i = blockIdx.x * blockDim.x + threadIdx.x;
    if (i >= Ktot * N) return;
    int k = i / N, n = i % N;
    float v = (k < fin) ? wx[(size_t)k * N + n] : wh[(size_t)(k - fin) * N + n];
    dst[bswz_idx(k, n, N / 16)] = (bf16_t)v;
}

// Stacked r-gate weight [wr ; a*wl0 ; a*wl1] -> swizzled bf16 (N = 64)
__global__ void sclstm_conv_wr(const float* __restrict__ wr,
                               const float* __restrict__ wl0,
                               const float* __restrict__ wl1,
                               bf16_t* __restrict__ dst, int fin, int layer)
{
    int Ktot = fin + layer * D_SZ;
    int i = blockIdx.x * blockDim.x + threadIdx.x;
    if (i >= Ktot * K_SZ) return;
    int k = i / K_SZ, n = i % K_SZ;
    float v;
    if (k < fin)                 v = wr[(size_t)k * K_SZ + n];
    else if (k < fin + D_SZ)     v = ALPHA_C * wl0[(size_t)(k - fin) * K_SZ + n];
    else                         v = ALPHA_C * wl1[(size_t)(k - fin - D_SZ) * K_SZ + n];
    dst[bswz_idx(k, n, K_SZ / 16)] = (bf16_t)v;
}

// Generic fp32 [K x N] row-major -> swizzled bf16 (projection weight)
__global__ void sclstm_conv_b(const float* __restrict__ src, bf16_t* __restrict__ dst,
                              int Kdim, int N)
{
    int i = blockIdx.x * blockDim.x + threadIdx.x;
    if (i >= Kdim * N) return;
    int k = i / N, n = i % N;
    dst[bswz_idx(k, n, N / 16)] = (bf16_t)src[i];
}

__global__ void sclstm_zero_bf16(bf16_t* p, int n)
{
    int i = blockIdx.x * blockDim.x + threadIdx.x;
    if (i < n) p[i] = (bf16_t)0.0f;
}
__global__ void sclstm_zero_f32(float* p, int n)
{
    int i = blockIdx.x * blockDim.x + threadIdx.x;
    if (i < n) p[i] = 0.0f;
}
__global__ void sclstm_init_d(float* d, const float* __restrict__ keys)
{
    int i = blockIdx.x * blockDim.x + threadIdx.x;
    if (i < 3 * B_SZ * K_SZ) d[i] = keys[i % (B_SZ * K_SZ)];
}
// Snapshot X columns [C, C+3D) (the h's) into Hprev (both swizzled)
__global__ void sclstm_copy_h(const bf16_t* __restrict__ X, bf16_t* __restrict__ Hp)
{
    int i = blockIdx.x * blockDim.x + threadIdx.x;
    const int total = (B_SZ / 16) * HCT * 512;
    if (i >= total) return;
    int elem = i & 511;
    int tile = i >> 9;
    int ct = tile % HCT;
    int rt = tile / HCT;
    Hp[i] = X[((size_t)(rt * XCT + (C_SZ / 32) + ct) << 9) + elem];
}

// ---------------------------------------------------------------------------
// Per-layer pointwise LSTM cell: r/d update, wd dot (K=64) via LDS, c/h update.
// One block per batch row, 512 threads (one per hidden dim).
__device__ __forceinline__ float sclstm_sigmoid(float x) { return 1.0f / (1.0f + __expf(-x)); }

__global__ __launch_bounds__(512)
void sclstm_pointwise(const float* __restrict__ gate, const float* __restrict__ gr,
                      const float* __restrict__ wxb, const float* __restrict__ whb,
                      float* __restrict__ c_st, float* __restrict__ d_st,
                      const float* __restrict__ wd, bf16_t* __restrict__ X, int xcol0)
{
    __shared__ float sdn[K_SZ];
    const int b = blockIdx.x;
    const int tid = threadIdx.x;

    if (tid < K_SZ) {
        float r  = sclstm_sigmoid(gr[b * K_SZ + tid]);
        float dn = r * d_st[b * K_SZ + tid];
        d_st[b * K_SZ + tid] = dn;
        sdn[tid] = dn;
    }
    __syncthreads();

    const float* g = gate + (size_t)b * 4 * D_SZ;
    const int dim = tid;
    float f  = sclstm_sigmoid(g[dim]              + wxb[dim]              + whb[dim]);
    float ii = sclstm_sigmoid(g[D_SZ + dim]       + wxb[D_SZ + dim]       + whb[D_SZ + dim]);
    float o  = sclstm_sigmoid(g[2 * D_SZ + dim]   + wxb[2 * D_SZ + dim]   + whb[2 * D_SZ + dim]);
    float cb = tanhf(         g[3 * D_SZ + dim]   + wxb[3 * D_SZ + dim]   + whb[3 * D_SZ + dim]);

    float acc = 0.0f;
#pragma unroll 8
    for (int k = 0; k < K_SZ; ++k) acc += sdn[k] * wd[k * D_SZ + dim];

    float c = f * c_st[(size_t)b * D_SZ + dim] + ii * cb + tanhf(acc);
    c_st[(size_t)b * D_SZ + dim] = c;
    float h = o * tanhf(c);
    X[xswz_idx(b, xcol0 + dim)] = (bf16_t)h;
}

// ---------------------------------------------------------------------------
extern "C" void kernel_launch(void* const* d_in, const int* in_sizes, int n_in,
                              void* d_out, int out_size, void* d_ws, size_t ws_size,
                              hipStream_t stream)
{
    (void)in_sizes; (void)n_in; (void)out_size; (void)ws_size;

    const float* inputs    = (const float*)d_in[0];
    const float* init_keys = (const float*)d_in[1];
    const float* wx_w[3] = {(const float*)d_in[2],  (const float*)d_in[8],  (const float*)d_in[14]};
    const float* wx_b[3] = {(const float*)d_in[3],  (const float*)d_in[9],  (const float*)d_in[15]};
    const float* wh_w[3] = {(const float*)d_in[4],  (const float*)d_in[10], (const float*)d_in[16]};
    const float* wh_b[3] = {(const float*)d_in[5],  (const float*)d_in[11], (const float*)d_in[17]};
    const float* wr_w[3] = {(const float*)d_in[6],  (const float*)d_in[12], (const float*)d_in[18]};
    const float* wl_w[3] = {(const float*)d_in[7],  (const float*)d_in[13], (const float*)d_in[19]};
    const float* wd      = (const float*)d_in[20];
    const float* proj_w  = (const float*)d_in[21];
    const float* proj_b  = (const float*)d_in[22];
    float* out = (float*)d_out;

    // Workspace carve-out (256B aligned). Total ~33 MB.
    char* ws = (char*)d_ws;
    size_t off = 0;
    auto carve = [&](size_t bytes) -> void* {
        void* p = ws + off;
        off = (off + bytes + 255) & ~(size_t)255;
        return p;
    };
    bf16_t* X  = (bf16_t*)carve((size_t)(B_SZ / 16) * XCT * 512 * 2);   // 2 MB
    bf16_t* Hp = (bf16_t*)carve((size_t)(B_SZ / 16) * HCT * 512 * 2);   // 1.5 MB
    bf16_t* Wg[3]; int fins[3];
    for (int l = 0; l < 3; ++l) {
        fins[l] = C_SZ + l * D_SZ;
        Wg[l] = (bf16_t*)carve((size_t)(fins[l] + D_SZ) * 4 * D_SZ * 2);
    }
    bf16_t* Wr[3];
    for (int l = 0; l < 3; ++l)
        Wr[l] = (bf16_t*)carve((size_t)(fins[l] + l * D_SZ) * K_SZ * 2);
    bf16_t* Pj   = (bf16_t*)carve((size_t)(3 * D_SZ) * OUT_SZ * 2);
    float*  gate = (float*)carve((size_t)B_SZ * 4 * D_SZ * 4);
    float*  gr   = (float*)carve((size_t)B_SZ * K_SZ * 4);
    float*  c_st = (float*)carve((size_t)3 * B_SZ * D_SZ * 4);
    float*  d_st = (float*)carve((size_t)3 * B_SZ * K_SZ * 4);

    const int TPB = 256;
    auto blocks = [](int n, int tpb) { return (n + tpb - 1) / tpb; };

    // ---- Weight conversion (bf16, pre-swizzled into WMMA B-fragment layout)
    for (int l = 0; l < 3; ++l) {
        int ng = (fins[l] + D_SZ) * 4 * D_SZ;
        sclstm_conv_gatew<<<blocks(ng, TPB), TPB, 0, stream>>>(wx_w[l], wh_w[l], Wg[l], fins[l]);
        int nr = (fins[l] + l * D_SZ) * K_SZ;
        sclstm_conv_wr<<<blocks(nr, TPB), TPB, 0, stream>>>(
            wr_w[l], wl_w[0], wl_w[1], Wr[l], fins[l], l);
    }
    sclstm_conv_b<<<blocks(3 * D_SZ * OUT_SZ, TPB), TPB, 0, stream>>>(proj_w, Pj, 3 * D_SZ, OUT_SZ);

    // ---- State init (every call; deterministic)
    sclstm_zero_bf16<<<blocks((B_SZ / 16) * XCT * 512, TPB), TPB, 0, stream>>>(
        X, (B_SZ / 16) * XCT * 512);
    sclstm_zero_bf16<<<blocks((B_SZ / 16) * HCT * 512, TPB), TPB, 0, stream>>>(
        Hp, (B_SZ / 16) * HCT * 512);
    sclstm_zero_f32<<<blocks(3 * B_SZ * D_SZ, TPB), TPB, 0, stream>>>(c_st, 3 * B_SZ * D_SZ);
    sclstm_init_d<<<blocks(3 * B_SZ * K_SZ, TPB), TPB, 0, stream>>>(d_st, init_keys);

    // ---- Time loop (sequential dependence; weights + activations stay in L2)
    for (int t = 0; t < T_SZ; ++t) {
        sclstm_conv_x<<<blocks(B_SZ * C_SZ, TPB), TPB, 0, stream>>>(inputs, X, t);

        for (int l = 0; l < 3; ++l) {
            // gate = X[:, 0:fin+D] @ [wx;wh]   (h_l columns still hold h^{t-1})
            int Kg = fins[l] + D_SZ;
            sclstm_gemm_big<<<dim3(4 * D_SZ / 128, B_SZ / 128), 256, 0, stream>>>(
                X, XCT, Wg[l], 4 * D_SZ / 16, gate, 4 * D_SZ, Kg);

            // gr = X[:, 0:fin] @ wr  +  sum_j a * hprev_j @ wl_j  (stacked K)
            sclstm_gemm_gr<<<dim3(B_SZ / 128), 256, 0, stream>>>(
                X, Hp, fins[l] / 32, l * (D_SZ / 32), Wr[l], gr);

            // pointwise cell update; writes new h_l (bf16) into X
            sclstm_pointwise<<<B_SZ, 512, 0, stream>>>(
                gate, gr, wx_b[l], wh_b[l],
                c_st + (size_t)l * B_SZ * D_SZ,
                d_st + (size_t)l * B_SZ * K_SZ,
                wd, X, C_SZ + l * D_SZ);
        }

        // out[:, t, :] = X[:, C:C+3D] @ proj_w + proj_b
        sclstm_gemm_main<<<dim3(OUT_SZ / 128, B_SZ / 64), 256, 0, stream>>>(
            X + ((size_t)(C_SZ / 32) << 9), XCT, Pj, OUT_SZ / 16,
            out + (size_t)t * OUT_SZ, T_SZ * OUT_SZ, proj_b, 3 * D_SZ);

        // snapshot h^{t} for next step's gr term
        sclstm_copy_h<<<blocks((B_SZ / 16) * HCT * 512, TPB), TPB, 0, stream>>>(X, Hp);
    }
}
